// GraphConvBlock_63702954934344
// MI455X (gfx1250) — compile-verified, matrix operand order checked by reading, not run on hardware
//
#include <hip/hip_runtime.h>
#include <hip/hip_bf16.h>

typedef __attribute__((ext_vector_type(16))) _Float16 v16h;
typedef __attribute__((ext_vector_type(8)))  float    v8f;

#define GN_EPS 1e-5f

// ---------------------------------------------------------------------------
// CSR row pointers from sorted edge_row via per-node binary search
// ---------------------------------------------------------------------------
__global__ void k_rowptr(const int* __restrict__ erow, int E, int V,
                         int* __restrict__ rp) {
  int v = blockIdx.x * blockDim.x + threadIdx.x;
  if (v > V) return;
  int lo = 0, hi = E;
  while (lo < hi) {
    int mid = (lo + hi) >> 1;
    if (erow[mid] < v) lo = mid + 1; else hi = mid;
  }
  rp[v] = lo;
}

// ---------------------------------------------------------------------------
// x [B,V,64] -> T0 [V, B*64]   (float4 granularity)
// ---------------------------------------------------------------------------
__global__ void k_x0(const float4* __restrict__ x, float4* __restrict__ T, int V) {
  int idx = blockIdx.x * 256 + threadIdx.x;          // over B*V*16
  int f4 = idx & 15;
  int v  = (idx >> 4) % V;
  int b  = (idx >> 4) / V;
  T[(size_t)v * 64 + b * 16 + f4] = x[idx];          // x flat index == idx
}

// ---------------------------------------------------------------------------
// SpMM: Tout = L*Tin (mode 0)  or  Tout = 2*L*Tin - Tout_old (mode 1, aliased
// with T_{k-2}; each element read once by the thread that rewrites it).
// One node per 64-thread block: row range and the edge stream (ew/ecol) are
// block-uniform -> lowered to scalar s_loads; the vector path carries only
// the coalesced 1KB gather lines of Tin.
// ---------------------------------------------------------------------------
__global__ void __launch_bounds__(64)
k_spmm(const float4* __restrict__ Tin, float4* __restrict__ Tout,
       const int* __restrict__ rp, const int* __restrict__ ecol,
       const float* __restrict__ ew, int mode) {
  int node = blockIdx.x;
  int t4   = threadIdx.x;            // 64 float4 columns = 256 floats
  int s = rp[node], e = rp[node + 1];
  float4 acc = {0.f, 0.f, 0.f, 0.f};
  for (int i = s; i < e; ++i) {
    float w = ew[i];                 // block-uniform -> s_load
    int   c = ecol[i];               // block-uniform -> s_load
    float4 xv = Tin[(size_t)c * 64 + t4];
    acc.x += w * xv.x; acc.y += w * xv.y;
    acc.z += w * xv.z; acc.w += w * xv.w;
  }
  size_t o = (size_t)node * 64 + t4;
  if (mode) {
    float4 p = Tout[o];
    acc.x = 2.f * acc.x - p.x; acc.y = 2.f * acc.y - p.y;
    acc.z = 2.f * acc.z - p.z; acc.w = 2.f * acc.w - p.w;
  }
  Tout[o] = acc;
}

// ---------------------------------------------------------------------------
// Projection accumulate: y[m,:] += T_k[m-row, :] @ W_k   via WMMA f16->f32.
// 16-bit fragment layout: lane<16 -> K packs {0..7,16..23}; lane>=16 -> K
// packs {8..15,24..31}; A row M = lane&15, B col N = lane&15 (+16*ntile).
// ---------------------------------------------------------------------------
#define WAVES 8
#define TPW   4
#define WPAD  68   // LDS row stride for transposed W (272B: 16B-aligned, bank-spread)

__device__ inline v16h pack16(float4 p0, float4 p1, float4 p2, float4 p3) {
  v16h f;
  f[0]  = (_Float16)p0.x; f[1]  = (_Float16)p0.y;
  f[2]  = (_Float16)p0.z; f[3]  = (_Float16)p0.w;
  f[4]  = (_Float16)p1.x; f[5]  = (_Float16)p1.y;
  f[6]  = (_Float16)p1.z; f[7]  = (_Float16)p1.w;
  f[8]  = (_Float16)p2.x; f[9]  = (_Float16)p2.y;
  f[10] = (_Float16)p2.z; f[11] = (_Float16)p2.w;
  f[12] = (_Float16)p3.x; f[13] = (_Float16)p3.y;
  f[14] = (_Float16)p3.z; f[15] = (_Float16)p3.w;
  return f;
}

template <int INIT>
__global__ void k_proj(const float* __restrict__ T, const float* __restrict__ Wk,
                       const float* __restrict__ bias, float* __restrict__ y,
                       int V) {
  __shared__ float Wl[64 * WPAD];                    // transposed: Wl[o][f]
  for (int i = threadIdx.x; i < 4096; i += 256) {
    int f = i >> 6, o = i & 63;
    Wl[o * WPAD + f] = Wk[i];
  }
  __syncthreads();

  const int lane   = threadIdx.x & 31;
  const int wave   = threadIdx.x >> 5;
  const int lrow   = lane & 15;
  const int kb     = (lane < 16) ? 0 : 8;
  const int rowoff = (lane < 16) ? 0 : 8;

  // B fragments: 4 N-tiles x 2 K-halves, loop-invariant for the wave
  v16h bf[4][2];
#pragma unroll
  for (int n = 0; n < 4; ++n) {
    const float* wc = &Wl[(n * 16 + lrow) * WPAD];
#pragma unroll
    for (int h = 0; h < 2; ++h) {
      const float* p = wc + h * 32 + kb;
      bf[n][h] = pack16(*(const float4*)(p),      *(const float4*)(p + 4),
                        *(const float4*)(p + 16), *(const float4*)(p + 20));
    }
  }

  const int tile0 = (blockIdx.x * WAVES + wave) * TPW;
#pragma unroll 1
  for (int ti = 0; ti < TPW; ++ti) {
    int t  = tile0 + ti;
    int m0 = t * 16;                  // V % 16 == 0 -> tiles never straddle b
    int b  = m0 / V;
    int v0 = m0 - b * V;
    const float* Trow = T + (size_t)(v0 + lrow) * 256 + b * 64;

    v16h a[2];
#pragma unroll
    for (int h = 0; h < 2; ++h) {
      const float* p = Trow + h * 32 + kb;
      a[h] = pack16(*(const float4*)(p),      *(const float4*)(p + 4),
                    *(const float4*)(p + 16), *(const float4*)(p + 20));
    }

    float* yrow = y + (size_t)(m0 + rowoff) * 64;
#pragma unroll
    for (int n = 0; n < 4; ++n) {
      int ncol = n * 16 + lrow;
      v8f c;
      if (INIT) {
        float bv = bias[ncol];
#pragma unroll
        for (int i = 0; i < 8; ++i) c[i] = bv;
      } else {
#pragma unroll
        for (int i = 0; i < 8; ++i) c[i] = yrow[i * 64 + ncol];
      }
      c = __builtin_amdgcn_wmma_f32_16x16x32_f16(false, a[0], false, bf[n][0],
                                                 (short)0, c, false, false);
      c = __builtin_amdgcn_wmma_f32_16x16x32_f16(false, a[1], false, bf[n][1],
                                                 (short)0, c, false, false);
#pragma unroll
      for (int i = 0; i < 8; ++i) yrow[i * 64 + ncol] = c[i];
    }
  }
}

// ---------------------------------------------------------------------------
// GroupNorm stats: per (batch, group) sum / sumsq over V*4 values
// ---------------------------------------------------------------------------
__global__ void k_zero(float* __restrict__ s) { s[threadIdx.x] = 0.f; }

__global__ void k_stats(const float4* __restrict__ y, float* __restrict__ stats,
                        int V) {
  __shared__ float ss[256], sq[256];
  int bg = blockIdx.y;
  int b  = bg >> 4;
  int g  = bg & 15;
  int v  = blockIdx.x * 256 + threadIdx.x;
  float4 p = y[(size_t)(b * V + v) * 16 + g];
  ss[threadIdx.x] = p.x + p.y + p.z + p.w;
  sq[threadIdx.x] = p.x * p.x + p.y * p.y + p.z * p.z + p.w * p.w;
  __syncthreads();
  for (int st = 128; st > 0; st >>= 1) {
    if (threadIdx.x < st) {
      ss[threadIdx.x] += ss[threadIdx.x + st];
      sq[threadIdx.x] += sq[threadIdx.x + st];
    }
    __syncthreads();
  }
  if (threadIdx.x == 0) {
    atomicAdd(&stats[bg * 2 + 0], ss[0]);
    atomicAdd(&stats[bg * 2 + 1], sq[0]);
  }
}

// ---------------------------------------------------------------------------
// Normalize + affine + ReLU in place (each float4 = one group's 4 channels)
// ---------------------------------------------------------------------------
__global__ void k_apply(float4* __restrict__ y, const float* __restrict__ stats,
                        const float* __restrict__ gamma,
                        const float* __restrict__ beta, int V) {
  int idx = blockIdx.x * 256 + threadIdx.x;          // over B*V*16
  int c4 = idx & 15;
  int b  = (idx >> 4) / V;
  int bg = b * 16 + c4;
  float n    = (float)V * 4.f;
  float mean = stats[bg * 2 + 0] / n;
  float var  = stats[bg * 2 + 1] / n - mean * mean;
  float rs   = rsqrtf(var + GN_EPS);
  float4 p = y[idx];
  int ch = c4 * 4;
  p.x = fmaxf(0.f, (p.x - mean) * rs * gamma[ch + 0] + beta[ch + 0]);
  p.y = fmaxf(0.f, (p.y - mean) * rs * gamma[ch + 1] + beta[ch + 1]);
  p.z = fmaxf(0.f, (p.z - mean) * rs * gamma[ch + 2] + beta[ch + 2]);
  p.w = fmaxf(0.f, (p.w - mean) * rs * gamma[ch + 3] + beta[ch + 3]);
  y[idx] = p;
}

// ---------------------------------------------------------------------------
extern "C" void kernel_launch(void* const* d_in, const int* in_sizes, int n_in,
                              void* d_out, int out_size, void* d_ws, size_t ws_size,
                              hipStream_t stream) {
  const float* x     = (const float*)d_in[0];
  const int*   erow  = (const int*)d_in[1];
  const int*   ecol  = (const int*)d_in[2];
  const float* ew    = (const float*)d_in[3];
  const float* wt    = (const float*)d_in[4];   // [5][64][64]
  const float* bias  = (const float*)d_in[5];
  const float* gamma = (const float*)d_in[6];
  const float* beta  = (const float*)d_in[7];
  float* y = (float*)d_out;

  const int V = 49152, B = 4;
  const int E = in_sizes[1];
  const size_t tbytes = (size_t)V * 256 * 4;    // one T buffer: 50.3 MB

  float* bufA  = (float*)d_ws;
  float* bufB  = (float*)((char*)d_ws + tbytes);
  int*   rp    = (int*)  ((char*)d_ws + 2 * tbytes);
  float* stats = (float*)((char*)d_ws + 2 * tbytes + 204800);

  dim3 blk(256);
  // CSR offsets + T0
  k_rowptr<<<(V + 1 + 255) / 256, blk, 0, stream>>>(erow, E, V, rp);
  k_x0<<<(B * V * 16) / 256, blk, 0, stream>>>((const float4*)x, (float4*)bufA, V);

  const int proj_blocks = (B * V / 16) / (WAVES * TPW);   // 384

  // k = 0 : y = T0 @ W0 + bias
  k_proj<1><<<proj_blocks, blk, 0, stream>>>(bufA, wt + 0 * 4096, bias, y, V);
  // T1 = L T0
  k_spmm<<<V, 64, 0, stream>>>((const float4*)bufA, (float4*)bufB,
                               rp, ecol, ew, 0);
  k_proj<0><<<proj_blocks, blk, 0, stream>>>(bufB, wt + 1 * 4096, bias, y, V);
  // T2 = 2 L T1 - T0   (into bufA, aliased with T0)
  k_spmm<<<V, 64, 0, stream>>>((const float4*)bufB, (float4*)bufA,
                               rp, ecol, ew, 1);
  k_proj<0><<<proj_blocks, blk, 0, stream>>>(bufA, wt + 2 * 4096, bias, y, V);
  // T3 = 2 L T2 - T1
  k_spmm<<<V, 64, 0, stream>>>((const float4*)bufA, (float4*)bufB,
                               rp, ecol, ew, 1);
  k_proj<0><<<proj_blocks, blk, 0, stream>>>(bufB, wt + 3 * 4096, bias, y, V);
  // T4 = 2 L T3 - T2
  k_spmm<<<V, 64, 0, stream>>>((const float4*)bufB, (float4*)bufA,
                               rp, ecol, ew, 1);
  k_proj<0><<<proj_blocks, blk, 0, stream>>>(bufA, wt + 4 * 4096, bias, y, V);

  // GroupNorm + ReLU
  k_zero<<<1, 128, 0, stream>>>(stats);
  dim3 sgrid(V / 256, B * 16);
  k_stats<<<sgrid, blk, 0, stream>>>((const float4*)y, stats, V);
  k_apply<<<(B * V * 16) / 256, blk, 0, stream>>>((float4*)y, stats, gamma, beta, V);
}
